// NormRNNResidual_60198261620836
// MI455X (gfx1250) — matrix operand order repaired and checked by reading, not compiled, not run on hardware
//
#include <hip/hip_runtime.h>

typedef __bf16 bf16;
typedef __attribute__((ext_vector_type(16))) __bf16 v16bf;
typedef __attribute__((ext_vector_type(8)))  float  v8f;
typedef __attribute__((ext_vector_type(4)))  unsigned int u32x4;
typedef __attribute__((ext_vector_type(8)))  int  i32x8;
typedef __attribute__((ext_vector_type(4)))  int  i32x4;

constexpr int B_ = 64;
constexpr int T_ = 4096;
constexpr int H_ = 64;
constexpr int G_ = 256;   // 4*H gates

// ---------------------------------------------------------------- helpers ---

__device__ __forceinline__ float sigmoidf_(float x) {
    return 1.0f / (1.0f + __expf(-x));
}
__device__ __forceinline__ float tanhf_(float x) {
    x = fminf(fmaxf(x, -15.0f), 15.0f);
    float e = __expf(2.0f * x);
    return (e - 1.0f) / (e + 1.0f);
}

__device__ __forceinline__ v8f wmma_bf16(v16bf a, v16bf b, v8f c) {
    return __builtin_amdgcn_wmma_f32_16x16x32_bf16(
        false, a, false, b, (short)0, c, false, false);
}

// A fragment (16x32 bf16, MxK), row-major source, row0 -> M=0 row, ld in elems.
// ISA 7.12.2: lanes 0-15 M=lane, elements {0..7,8..15} -> K {klo..klo+7, 16+klo..},
// klo = 0 (lanes 0-15) / 8 (lanes 16-31).
__device__ __forceinline__ v16bf load_A_bf16(const bf16* row0, int ld, int kb, int lane) {
    int m   = lane & 15;
    int klo = (lane < 16) ? 0 : 8;
    const bf16* row = row0 + (long)m * ld;
    union { v16bf v; bf16 e[16]; } a;
#pragma unroll
    for (int i = 0; i < 8; ++i) {
        a.e[i]     = row[kb + klo + i];
        a.e[i + 8] = row[kb + 16 + klo + i];
    }
    return a.v;
}

// A fragment from f32 with fused instance-norm: (x-mu)*rstd -> bf16.
__device__ __forceinline__ v16bf load_A_norm(const float* row0, int ld,
                                             const float* mu, const float* rs,
                                             int kb, int lane) {
    int m   = lane & 15;
    int klo = (lane < 16) ? 0 : 8;
    const float* row = row0 + (long)m * ld;
    union { v16bf v; bf16 e[16]; } a;
#pragma unroll
    for (int i = 0; i < 8; ++i) {
        int k0 = kb + klo + i;
        int k1 = kb + 16 + klo + i;
        a.e[i]     = (bf16)((row[k0] - mu[k0]) * rs[k0]);
        a.e[i + 8] = (bf16)((row[k1] - mu[k1]) * rs[k1]);
    }
    return a.v;
}

// B fragment (32x16 bf16, KxN) from W[g,f] row-major: column n of B == row g of W
// (contiguous). Lane holds n = lane&15, K = kb + (lane<16?0:8) + (i<8 ? i : i+8).
__device__ __forceinline__ v16bf load_B_bf16(const bf16* W, int gbase, int kb,
                                             int ld, int lane) {
    int n  = lane & 15;
    int k0 = kb + ((lane < 16) ? 0 : 8);
    const bf16* p = W + (long)(gbase + n) * ld + k0;
    union { v16bf v; bf16 e[16]; } b;
#pragma unroll
    for (int i = 0; i < 16; ++i) b.e[i] = p[(i < 8) ? i : i + 8];
    return b.v;
}

// Same, but converting from f32 source (used once to pin Whh tiles in VGPRs).
__device__ __forceinline__ v16bf load_B_f32cvt(const float* W, int gbase, int kb,
                                               int ld, int lane) {
    int n  = lane & 15;
    int k0 = kb + ((lane < 16) ? 0 : 8);
    const float* p = W + (long)(gbase + n) * ld + k0;
    union { v16bf v; bf16 e[16]; } b;
#pragma unroll
    for (int i = 0; i < 16; ++i) b.e[i] = (bf16)p[(i < 8) ? i : i + 8];
    return b.v;
}

// --------------------------------------------------------------- TDM issue ---
// 1-row tile of `nelem` f32 elements, contiguous global -> contiguous LDS.
// D# per CDNA5 ISA (group0: count=1,type=2; group1:
// data_size=4B, tensor_dim0=tile_dim0=nelem, tile_dim1=tensor_dim1=1).
// clang-23 toolchain: 6-arg builtin (g0, g1, g2, g3, extra i32x8, cpol).
__device__ __forceinline__ void tdm_load_tile(const float* gsrc, float* ldst, int nelem) {
    unsigned long long ga = (unsigned long long)(uintptr_t)gsrc;
    unsigned int la = (unsigned int)(uintptr_t)ldst;   // flat LDS addr[31:0] = offset
    u32x4 g0;
    g0[0] = 1u;                                         // count=1 valid descriptor
    g0[1] = la;                                         // lds_addr
    g0[2] = (unsigned int)(ga & 0xffffffffu);           // global_addr[31:0]
    g0[3] = (unsigned int)((ga >> 32) & 0x01ffffffu)    // global_addr[56:32]
          | 0x80000000u;                                // type=2 ("image")
    i32x8 g1;
    g1[0] = 0x00020000;                                 // workgroup_mask=0, data_size=4B
    g1[1] = (int)((nelem & 0xffff) << 16);              // tensor_dim0[15:0]
    g1[2] = (int)(((unsigned)nelem >> 16) & 0xffffu) | (1 << 16); // td0[31:16], tensor_dim1=1
    g1[3] = (int)((nelem & 0xffff) << 16);              // tile_dim0
    g1[4] = 1;                                          // tile_dim1=1, tile_dim2=0
    g1[5] = nelem;                                      // tensor_dim0_stride[31:0]
    g1[6] = 0;
    g1[7] = 0;
    i32x4 z4 = {0, 0, 0, 0};                            // groups 2/3 unused (<=2D)
    i32x8 z8 = {0, 0, 0, 0, 0, 0, 0, 0};
    __builtin_amdgcn_tensor_load_to_lds(g0, g1, z4, z4, z8, 0);
}

// ------------------------------------------------------------ stats kernel ---
__global__ void stats_kernel(const float* __restrict__ x,
                             float* __restrict__ mu, float* __restrict__ rstd) {
    __shared__ float ssum[16][64];
    __shared__ float ssq[16][64];
    int b  = blockIdx.x;
    int f  = threadIdx.x & 63;
    int sl = threadIdx.x >> 6;
    const float* p = x + ((long)b * T_ + (long)sl * 256) * H_ + f;
    float s = 0.f, q = 0.f;
    for (int t = 0; t < 256; ++t) {
        float v = p[(long)t * H_];
        s += v; q += v * v;
    }
    ssum[sl][f] = s; ssq[sl][f] = q;
    __syncthreads();
    if (sl == 0) {
#pragma unroll
        for (int i = 1; i < 16; ++i) { s += ssum[i][f]; q += ssq[i][f]; }
        float m   = s * (1.0f / T_);
        float var = q * (1.0f / T_) - m * m;
        mu[b * H_ + f]   = m;
        rstd[b * H_ + f] = rsqrtf(var + 1e-5f);
    }
}

// ------------------------------------------------------- weight prep kernel ---
__global__ void prep_kernel(const float* Wih0, const float* Wih1, const float* Wfc,
                            const float* bih0, const float* bhh0,
                            const float* bih1, const float* bhh1,
                            bf16* Wih0b, bf16* Wih1b, bf16* Wfcb,
                            float* bias0, float* bias1) {
    int i = blockIdx.x * blockDim.x + threadIdx.x;
    if (i < G_ * H_) { Wih0b[i] = (bf16)Wih0[i]; Wih1b[i] = (bf16)Wih1[i]; }
    if (i < H_ * H_) Wfcb[i] = (bf16)Wfc[i];
    if (i < G_)      { bias0[i] = bih0[i] + bhh0[i]; bias1[i] = bih1[i] + bhh1[i]; }
}

// ----------------------------------------------------------- x-projection ---
// NORM:  A = instance-norm(x[B,T,H]) tile      (layer 0)
// !NORM: A = h[T,B,H] bf16 tile                (layer 1)
// Output xp[T,B,G]: per-time-step tiles contiguous for the recurrence/TDM.
// mtile = b*(T/16) + t0/16; one M-tile per wave, wave loops 16 N-tiles.
template <bool NORM>
__global__ void xproj_kernel(const float* __restrict__ xf, const bf16* __restrict__ xb,
                             const float* __restrict__ mu, const float* __restrict__ rstd,
                             const bf16* __restrict__ W, const float* __restrict__ bias,
                             float* __restrict__ xp) {
    int wave = threadIdx.x >> 5;
    int lane = threadIdx.x & 31;
    long mtile = (long)blockIdx.x * 8 + wave;   // 16384 tiles
    int  b  = (int)(mtile >> 8);                // T/16 = 256 tiles per batch
    int  t0 = ((int)mtile & 255) * 16;
    v16bf a0, a1;
    if (NORM) {
        const float* row0 = xf + ((long)b * T_ + t0) * H_;
        const float* mub  = mu + b * H_;
        const float* rsb  = rstd + b * H_;
        a0 = load_A_norm(row0, H_, mub, rsb, 0,  lane);
        a1 = load_A_norm(row0, H_, mub, rsb, 32, lane);
    } else {
        const bf16* row0 = xb + ((long)t0 * B_ + b) * H_;
        a0 = load_A_bf16(row0, B_ * H_, 0,  lane);
        a1 = load_A_bf16(row0, B_ * H_, 32, lane);
    }
    int n  = lane & 15;
    int bb = (lane < 16) ? 0 : 8;
#pragma unroll 4
    for (int g = 0; g < 16; ++g) {
        v16bf b0 = load_B_bf16(W, g * 16, 0,  H_, lane);
        v16bf b1 = load_B_bf16(W, g * 16, 32, H_, lane);
        v8f acc = {};
        acc = wmma_bf16(a0, b0, acc);
        acc = wmma_bf16(a1, b1, acc);
        float bs = bias[g * 16 + n];
        union { v8f v; float e[8]; } A; A.v = acc;
#pragma unroll
        for (int j = 0; j < 8; ++j)
            xp[((long)(t0 + bb + j) * B_ + b) * G_ + g * 16 + n] = A.e[j] + bs;
    }
}

// ------------------------------------------------------------ LSTM recurrence ---
// 4 WGs x 16 batches x 16 waves. Whh B-fragments pinned in VGPRs (fixed per wave).
// xp[t] tiles (16KB contiguous) streamed into double-buffered LDS by the Tensor
// Data Mover (wave 0 issues; s_wait_tensorcnt + WG barrier publishes). Gate
// preactivations are written in place into the consumed xp LDS tile.
__global__ __launch_bounds__(512)
void lstm_kernel(const float* __restrict__ xp,   // [T,B,256] f32
                 const float* __restrict__ Whh,  // [256,64]  f32
                 bf16* __restrict__ h_out) {     // [T,B,64]  bf16
    __shared__ float xpL[2][16 * G_];   // 2 x 16 KB
    __shared__ bf16  hbuf[16 * H_];     // 2 KB

    int tid   = threadIdx.x;
    int wave  = tid >> 5;
    int lane  = tid & 31;
    int bbase = blockIdx.x * 16;
    int gbase = wave * 16;
    int n  = lane & 15;
    int bb = (lane < 16) ? 0 : 8;

    // Whh tile for this wave, kept in registers for all 4096 steps.
    v16bf wb0 = load_B_f32cvt(Whh, gbase, 0,  H_, lane);
    v16bf wb1 = load_B_f32cvt(Whh, gbase, 32, H_, lane);

    for (int i = tid; i < 16 * H_; i += 512) hbuf[i] = (bf16)0.0f;
    if (wave == 0) {
        tdm_load_tile(xp + (long)bbase * G_, xpL[0], 16 * G_);
        __builtin_amdgcn_s_wait_tensorcnt(0);
    }
    __syncthreads();

    v8f c_state = {};                   // waves 0-3 only
    int cur = 0;
    for (int t = 0; t < T_; ++t) {
        float* gxp = xpL[cur];

        // recurrent matmul: per-wave 16x16 gate tile
        v16bf a0 = load_A_bf16(hbuf, H_, 0,  lane);
        v16bf a1 = load_A_bf16(hbuf, H_, 32, lane);
        v8f acc = {};
        acc = wmma_bf16(a0, wb0, acc);
        acc = wmma_bf16(a1, wb1, acc);

        // preactivation = WMMA + xp (in-place into consumed LDS tile)
        union { v8f v; float e[8]; } A; A.v = acc;
#pragma unroll
        for (int j = 0; j < 8; ++j) {
            int a = (bb + j) * G_ + gbase + n;
            gxp[a] = A.e[j] + gxp[a];
        }

        // kick TDM for step t+1 into the other buffer (overlaps update phase)
        if (wave == 0 && t + 1 < T_)
            tdm_load_tile(xp + ((long)(t + 1) * B_ + bbase) * G_,
                          xpL[cur ^ 1], 16 * G_);
        __syncthreads();

        // elementwise cell update: wave w<4 owns hidden units 16w..16w+15
        if (wave < 4) {
            int h = gbase + n;
            union { v8f v; float e[8]; } C; C.v = c_state;
#pragma unroll
            for (int j = 0; j < 8; ++j) {
                int b = bb + j;
                const float* gr = &gxp[b * G_];
                float ig = sigmoidf_(gr[h]);
                float fg = sigmoidf_(gr[64 + h]);
                float gg = tanhf_(gr[128 + h]);
                float og = sigmoidf_(gr[192 + h]);
                float c  = fg * C.e[j] + ig * gg;
                C.e[j]   = c;
                float hv = og * tanhf_(c);
                bf16  hb = (bf16)hv;
                hbuf[b * H_ + h] = hb;
                h_out[((long)t * B_ + bbase + b) * H_ + h] = hb;
            }
            c_state = C.v;
        }
        if (wave == 0) __builtin_amdgcn_s_wait_tensorcnt(0);
        __syncthreads();
        cur ^= 1;
    }
}

// ------------------------------------------------------------- FC + residual ---
__global__ void fc_kernel(const bf16* __restrict__ h1,   // [T,B,64]
                          const bf16* __restrict__ Wfc, const float* __restrict__ bfc,
                          const float* __restrict__ x,   // [B,T,64]
                          float* __restrict__ out) {     // [B,T,64]
    int wave = threadIdx.x >> 5;
    int lane = threadIdx.x & 31;
    long mtile = (long)blockIdx.x * 8 + wave;
    int  b  = (int)(mtile >> 8);
    int  t0 = ((int)mtile & 255) * 16;
    const bf16* row0 = h1 + ((long)t0 * B_ + b) * H_;
    v16bf a0 = load_A_bf16(row0, B_ * H_, 0,  lane);
    v16bf a1 = load_A_bf16(row0, B_ * H_, 32, lane);
    int n  = lane & 15;
    int bb = (lane < 16) ? 0 : 8;
#pragma unroll
    for (int g = 0; g < 4; ++g) {
        v16bf b0 = load_B_bf16(Wfc, g * 16, 0,  H_, lane);
        v16bf b1 = load_B_bf16(Wfc, g * 16, 32, H_, lane);
        v8f acc = {};
        acc = wmma_bf16(a0, b0, acc);
        acc = wmma_bf16(a1, b1, acc);
        float bs = bfc[g * 16 + n];
        union { v8f v; float e[8]; } A; A.v = acc;
#pragma unroll
        for (int j = 0; j < 8; ++j) {
            long idx = ((long)b * T_ + t0 + bb + j) * H_ + g * 16 + n;
            out[idx] = A.e[j] + bs + x[idx];
        }
    }
}

// ---------------------------------------------------------------- launcher ---

static inline size_t align256(size_t v) { return (v + 255) & ~size_t(255); }

extern "C" void kernel_launch(void* const* d_in, const int* in_sizes, int n_in,
                              void* d_out, int out_size, void* d_ws, size_t ws_size,
                              hipStream_t stream) {
    const float* x    = (const float*)d_in[0];
    const float* Wih0 = (const float*)d_in[1];
    const float* Whh0 = (const float*)d_in[2];
    const float* bih0 = (const float*)d_in[3];
    const float* bhh0 = (const float*)d_in[4];
    const float* Wih1 = (const float*)d_in[5];
    const float* Whh1 = (const float*)d_in[6];
    const float* bih1 = (const float*)d_in[7];
    const float* bhh1 = (const float*)d_in[8];
    const float* Wfc  = (const float*)d_in[9];
    const float* bfc  = (const float*)d_in[10];
    float* out = (float*)d_out;

    char* ws = (char*)d_ws;
    size_t off = 0;
    float* mu    = (float*)(ws + off); off = align256(off + B_ * H_ * 4);
    float* rstd  = (float*)(ws + off); off = align256(off + B_ * H_ * 4);
    bf16*  Wih0b = (bf16*)(ws + off);  off = align256(off + G_ * H_ * 2);
    bf16*  Wih1b = (bf16*)(ws + off);  off = align256(off + G_ * H_ * 2);
    bf16*  Wfcb  = (bf16*)(ws + off);  off = align256(off + H_ * H_ * 2);
    float* bias0 = (float*)(ws + off); off = align256(off + G_ * 4);
    float* bias1 = (float*)(ws + off); off = align256(off + G_ * 4);
    bf16*  h0    = (bf16*)(ws + off);  off = align256(off + (size_t)B_ * T_ * H_ * 2);
    bf16*  h1    = (bf16*)(ws + off);  off = align256(off + (size_t)B_ * T_ * H_ * 2);
    float* xp    = (float*)(ws + off); off = align256(off + (size_t)B_ * T_ * G_ * 4);
    (void)ws_size; (void)n_in; (void)in_sizes; (void)out_size;

    stats_kernel<<<B_, 1024, 0, stream>>>(x, mu, rstd);

    prep_kernel<<<(G_ * H_ + 255) / 256, 256, 0, stream>>>(
        Wih0, Wih1, Wfc, bih0, bhh0, bih1, bhh1,
        Wih0b, Wih1b, Wfcb, bias0, bias1);

    const int gemm_blocks = (B_ * T_ / 16) / 8;   // 2048

    xproj_kernel<true><<<gemm_blocks, 256, 0, stream>>>(
        x, nullptr, mu, rstd, Wih0b, bias0, xp);

    lstm_kernel<<<B_ / 16, 512, 0, stream>>>(xp, Whh0, h0);

    xproj_kernel<false><<<gemm_blocks, 256, 0, stream>>>(
        nullptr, h0, nullptr, nullptr, Wih1b, bias1, xp);

    lstm_kernel<<<B_ / 16, 512, 0, stream>>>(xp, Whh1, h1);

    fc_kernel<<<gemm_blocks, 256, 0, stream>>>(h1, Wfcb, bfc, x, out);
}
